// SelfAttention_V2_2224793059596
// MI455X (gfx1250) — compile-verified, matrix operand order checked by reading, not run on hardware
//
#include <hip/hip_runtime.h>

typedef __attribute__((ext_vector_type(16))) _Float16 v16h;
typedef __attribute__((ext_vector_type(8)))  _Float16 v8h;
typedef __attribute__((ext_vector_type(8)))  float    v8f;
typedef __attribute__((ext_vector_type(4)))  float    v4f;

#define N_TOK 8192
#define DIM   1024
#define SOFT_SCALE 0.03125f   // 1/sqrt(1024)
#define KB 64                 // keys per flash-attention iteration

// ---- workspace layout (bytes) ----
#define OFF_XH  (0)                               // 16 MB
#define OFF_WQ  (OFF_XH + (size_t)N_TOK*DIM*2)    // 2 MB
#define OFF_WK  (OFF_WQ + (size_t)DIM*DIM*2)
#define OFF_WV  (OFF_WK + (size_t)DIM*DIM*2)
#define OFF_Q   (OFF_WV + (size_t)DIM*DIM*2)      // 16 MB
#define OFF_K   (OFF_Q  + (size_t)N_TOK*DIM*2)    // 16 MB
#define OFF_VT  (OFF_K  + (size_t)N_TOK*DIM*2)    // 16 MB (V transposed [DIM][N_TOK])

__device__ __forceinline__ v8f wmma_f16(v16h a, v16h b, v8f c) {
    return __builtin_amdgcn_wmma_f32_16x16x32_f16(
        false, a, false, b, (short)0, c, false, false);
}

// A-layout (and mirrored B-layout) 16x32 f16 fragment from row-major M[ld].
__device__ __forceinline__ v16h load_frag(const _Float16* __restrict__ M,
                                          int r0, int c0, int ld, int lane) {
    int row = r0 + (lane & 15);
    int col = c0 + ((lane >> 4) << 3);
    const _Float16* p = M + (size_t)row * ld + col;
    union { v16h f; v8h h[2]; } u;
    u.h[0] = *(const v8h*)p;
    u.h[1] = *(const v8h*)(p + 16);
    return u.f;
}

// ---------------- f32 -> f16 convert, 8 elems/thread ----------------
__global__ void cvt_f16_kernel(const float* __restrict__ in,
                               _Float16* __restrict__ out, int n8) {
    int i = blockIdx.x * blockDim.x + threadIdx.x;
    if (i >= n8) return;
    v4f a = *(const v4f*)(in + (size_t)i * 8);
    v4f b = *(const v4f*)(in + (size_t)i * 8 + 4);
    v8h h;
#pragma unroll
    for (int e = 0; e < 4; ++e) { h[e] = (_Float16)a[e]; h[4 + e] = (_Float16)b[e]; }
    *(v8h*)(out + (size_t)i * 8) = h;
}

// ---------------- QKV projection: Y = Xh @ W^T ----------------
// grid: (N_TOK/16, DIM/512); block 256 (8 waves). Wave w owns a 16x64 strip.
// Fully unrolled k-loop with one-step prefetch so loads overlap WMMAs.
__global__ __launch_bounds__(256)
void gemm_qkv_kernel(const _Float16* __restrict__ X,
                     const _Float16* __restrict__ W,
                     _Float16* __restrict__ Y, int transOut) {
    int wave = threadIdx.x >> 5;
    int lane = threadIdx.x & 31;
    int i0 = blockIdx.x * 16;
    int o0 = blockIdx.y * 512 + wave * 64;

    v8f acc[4];
#pragma unroll
    for (int t = 0; t < 4; ++t) acc[t] = (v8f){};

    // prologue: fragments for k = 0
    v16h a_c = load_frag(X, i0, 0, DIM, lane);
    v16h b_c[4];
#pragma unroll
    for (int t = 0; t < 4; ++t) b_c[t] = load_frag(W, o0 + t * 16, 0, DIM, lane);

#pragma unroll
    for (int k = 0; k < DIM; k += 32) {
        v16h a_n;
        v16h b_n[4];
        if (k + 32 < DIM) {                   // compile-time after full unroll
            a_n = load_frag(X, i0, k + 32, DIM, lane);
#pragma unroll
            for (int t = 0; t < 4; ++t)
                b_n[t] = load_frag(W, o0 + t * 16, k + 32, DIM, lane);
        }
#pragma unroll
        for (int t = 0; t < 4; ++t) acc[t] = wmma_f16(a_c, b_c[t], acc[t]);
        a_c = a_n;
#pragma unroll
        for (int t = 0; t < 4; ++t) b_c[t] = b_n[t];
    }

    int colc = lane & 15;
    int rbase = (lane >> 4) << 3;
#pragma unroll
    for (int t = 0; t < 4; ++t) {
#pragma unroll
        for (int r = 0; r < 8; ++r) {
            int row = i0 + r + rbase;                 // token index
            int col = o0 + t * 16 + colc;             // output-dim index
            _Float16 v = (_Float16)acc[t][r];
            if (transOut) Y[(size_t)col * N_TOK + row] = v;   // Vt[d][token]
            else          Y[(size_t)row * DIM + col]   = v;   // Q/K [token][d]
        }
    }
}

// ---------------- flash attention ----------------
// grid: N_TOK/16 blocks; block 256 (8 waves); 16 query rows per block.
// Waves split the 1024-d reduction for S (128 dims each -> LDS reduce) and
// split the 1024-d output (128 dims each) for O += P @ V. 64 keys/iteration.
__global__ __launch_bounds__(256)
void flash_attn_kernel(const _Float16* __restrict__ Q,
                       const _Float16* __restrict__ K,
                       const _Float16* __restrict__ Vt,
                       float* __restrict__ Out) {
    __shared__ float s_part[8][16 * KB];   // per-wave partial S (4 tiles of 16x16)
    __shared__ float s_final[16 * KB];     // reduced S, [row][key 0..63]
    __shared__ float m_arr[16], c_arr[16], l_arr[16];

    int wave = threadIdx.x >> 5;
    int lane = threadIdx.x & 31;
    int i0 = blockIdx.x * 16;
    int d0 = wave * 128;

    v16h qf[4];
#pragma unroll
    for (int s = 0; s < 4; ++s)
        qf[s] = load_frag(Q, i0, d0 + 32 * s, DIM, lane);

    v8f oacc[8];
#pragma unroll
    for (int t = 0; t < 8; ++t) oacc[t] = (v8f){};

    if (threadIdx.x < 16) { m_arr[threadIdx.x] = -1e30f; l_arr[threadIdx.x] = 0.f; }
    __syncthreads();

    int colc = lane & 15;
    int rbase = (lane >> 4) << 3;

    for (int j0 = 0; j0 < N_TOK; j0 += KB) {
        // ---- partial S = Q_slice @ K_slice^T (four 16x16 key tiles) ----
        // one-step prefetch across the 4 k-steps of the dim slice
        v8f sacc[4];
#pragma unroll
        for (int kt = 0; kt < 4; ++kt) sacc[kt] = (v8f){};

        v16h kf_c[4];
#pragma unroll
        for (int kt = 0; kt < 4; ++kt)
            kf_c[kt] = load_frag(K, j0 + kt * 16, d0, DIM, lane);

#pragma unroll
        for (int s = 0; s < 4; ++s) {
            v16h kf_n[4];
            if (s < 3) {
#pragma unroll
                for (int kt = 0; kt < 4; ++kt)
                    kf_n[kt] = load_frag(K, j0 + kt * 16, d0 + 32 * (s + 1), DIM, lane);
            }
#pragma unroll
            for (int kt = 0; kt < 4; ++kt)
                sacc[kt] = wmma_f16(qf[s], kf_c[kt], sacc[kt]);
#pragma unroll
            for (int kt = 0; kt < 4; ++kt) kf_c[kt] = kf_n[kt];
        }
#pragma unroll
        for (int kt = 0; kt < 4; ++kt)
#pragma unroll
            for (int r = 0; r < 8; ++r)
                s_part[wave][kt * 256 + (r + rbase) * 16 + colc] = sacc[kt][r];
        __syncthreads();

        // ---- cross-wave reduce + scale: 256 threads x 4 elements ----
#pragma unroll
        for (int q = 0; q < 4; ++q) {
            int e = threadIdx.x + q * 256;
            float v = 0.f;
#pragma unroll
            for (int w = 0; w < 8; ++w) v += s_part[w][e];
            int kt = e >> 8, m = (e >> 4) & 15, n = e & 15;
            s_final[m * KB + kt * 16 + n] = v * SOFT_SCALE;
        }
        __syncthreads();

        // ---- online softmax stats (one thread per query row) ----
        if (threadIdx.x < 16) {
            int row = threadIdx.x;
            float mo = m_arr[row], lo = l_arr[row];
            float mx = mo;
            for (int j = 0; j < KB; ++j) mx = fmaxf(mx, s_final[row * KB + j]);
            float c = __expf(mo - mx);
            float ls = 0.f;
            for (int j = 0; j < KB; ++j) ls += __expf(s_final[row * KB + j] - mx);
            m_arr[row] = mx; c_arr[row] = c; l_arr[row] = c * lo + ls;
        }
        __syncthreads();

        // ---- rescale running output accumulators ----
#pragma unroll
        for (int r = 0; r < 8; ++r) {
            float c = c_arr[r + rbase];
#pragma unroll
            for (int t = 0; t < 8; ++t) oacc[t][r] *= c;
        }

        // ---- build P as two f16 A-fragments (16 rows x 2x32 keys) ----
        v16h pf[2];
        {
            int prow = lane & 15;
            int pc0 = (lane >> 4) << 3;
            float mrow = m_arr[prow];
#pragma unroll
            for (int h = 0; h < 2; ++h) {
                int base = prow * KB + h * 32 + pc0;
#pragma unroll
                for (int e = 0; e < 8; ++e)
                    pf[h][e] = (_Float16)__expf(s_final[base + e] - mrow);
#pragma unroll
                for (int e = 0; e < 8; ++e)
                    pf[h][8 + e] = (_Float16)__expf(s_final[base + 16 + e] - mrow);
            }
        }

        // ---- O_slice += P @ V_slice, one-step prefetch over 16 (t,h) steps ----
        {
            v16h vf_c = load_frag(Vt, d0, j0, N_TOK, lane);
#pragma unroll
            for (int i = 0; i < 16; ++i) {
                int t = i >> 1, h = i & 1;
                v16h vf_n;
                if (i < 15) {
                    int tn = (i + 1) >> 1, hn = (i + 1) & 1;
                    vf_n = load_frag(Vt, d0 + tn * 16, j0 + hn * 32, N_TOK, lane);
                }
                oacc[t] = wmma_f16(pf[h], vf_c, oacc[t]);
                vf_c = vf_n;
            }
        }
        // next iteration's first barrier protects s_final/m/c/l reuse
    }

    // ---- epilogue ----
#pragma unroll
    for (int r = 0; r < 8; ++r) {
        int row = r + rbase;
        float inv = 1.0f / l_arr[row];
#pragma unroll
        for (int t = 0; t < 8; ++t)
            Out[(size_t)(i0 + row) * DIM + d0 + t * 16 + colc] = oacc[t][r] * inv;
    }
}

extern "C" void kernel_launch(void* const* d_in, const int* in_sizes, int n_in,
                              void* d_out, int out_size, void* d_ws, size_t ws_size,
                              hipStream_t stream) {
    const float* x  = (const float*)d_in[0];
    const float* Wq = (const float*)d_in[1];
    const float* Wk = (const float*)d_in[2];
    const float* Wv = (const float*)d_in[3];
    float* out = (float*)d_out;

    char* ws = (char*)d_ws;
    _Float16* xh  = (_Float16*)(ws + OFF_XH);
    _Float16* wqh = (_Float16*)(ws + OFF_WQ);
    _Float16* wkh = (_Float16*)(ws + OFF_WK);
    _Float16* wvh = (_Float16*)(ws + OFF_WV);
    _Float16* Qh  = (_Float16*)(ws + OFF_Q);
    _Float16* Kh  = (_Float16*)(ws + OFF_K);
    _Float16* Vt  = (_Float16*)(ws + OFF_VT);

    const int nx8 = N_TOK * DIM / 8;
    const int nw8 = DIM * DIM / 8;

    cvt_f16_kernel<<<(nx8 + 255) / 256, 256, 0, stream>>>(x,  xh,  nx8);
    cvt_f16_kernel<<<(nw8 + 255) / 256, 256, 0, stream>>>(Wq, wqh, nw8);
    cvt_f16_kernel<<<(nw8 + 255) / 256, 256, 0, stream>>>(Wk, wkh, nw8);
    cvt_f16_kernel<<<(nw8 + 255) / 256, 256, 0, stream>>>(Wv, wvh, nw8);

    dim3 ggrid(N_TOK / 16, DIM / 512);   // 512 x 2
    gemm_qkv_kernel<<<ggrid, 256, 0, stream>>>(xh, wqh, Qh, 0);
    gemm_qkv_kernel<<<ggrid, 256, 0, stream>>>(xh, wkh, Kh, 0);
    gemm_qkv_kernel<<<ggrid, 256, 0, stream>>>(xh, wvh, Vt, 1);  // transposed

    flash_attn_kernel<<<N_TOK / 16, 256, 0, stream>>>(Qh, Kh, Vt, out);
}
